// MultiHeadAttention_LSTM_49340584297059
// MI455X (gfx1250) — matrix operand order repaired and checked by reading, not compiled, not run on hardware
//
#include <hip/hip_runtime.h>
#include <hip/hip_bf16.h>

// MI455X / gfx1250, wave32. All matrix math on v_wmma_f32_16x16x32_bf16.

typedef __attribute__((ext_vector_type(16))) __bf16 v16bf;
typedef __attribute__((ext_vector_type(8)))  float  v8f;

#define Bq   256
#define Gq   100
#define GT   112   // 7*16  : padded M extent
#define GP   128   // 8*16  : padded K extent
#define IND  128
#define KD   32
#define NORMQK 0.17677669529663687f   // 1/sqrt(32)
#define LSTM_PF 32                    // prefetch distance in steps (32B/step/wave, 8KB stride)

union Frag { v16bf v; unsigned short u[16]; uint4 q[2]; };

__device__ __forceinline__ unsigned short f2bf(float f) {
  unsigned int x = __builtin_bit_cast(unsigned int, f);
  x = (x + 0x7FFFu + ((x >> 16) & 1u)) >> 16;   // RNE
  return (unsigned short)x;
}

// A operand: 16(M) x 32(K), row-major source with leading dim ld.
// lane<16: row m0+lane, K [k0..k0+15]
// lane>=16: row m0+lane-16, K [k0+16..k0+31]
__device__ __forceinline__ v16bf load_a(const unsigned short* base, int ld,
                                        int m0, int k0, int lane) {
  int r  = m0 + (lane & 15);
  int kb = k0 + ((lane >> 4) << 4);
  const unsigned short* p = base + (size_t)r * ld + kb;
  Frag f;
  f.q[0] = *reinterpret_cast<const uint4*>(p);
  f.q[1] = *reinterpret_cast<const uint4*>(p + 8);
  return f.v;
}

// B operand: 32(K) x 16(N), loaded from an N-major array bt[N][K] (leading dim ld).
// lane<16: col n0+lane, K [k0..k0+15]; lane>=16: col n0+lane-16, K [k0+16..k0+31]
__device__ __forceinline__ v16bf load_bt(const unsigned short* bt, int ld,
                                         int n0, int k0, int lane) {
  int c  = n0 + (lane & 15);
  int kb = k0 + ((lane >> 4) << 4);
  const unsigned short* p = bt + (size_t)c * ld + kb;
  Frag f;
  f.q[0] = *reinterpret_cast<const uint4*>(p);
  f.q[1] = *reinterpret_cast<const uint4*>(p + 8);
  return f.v;
}

#define WMMA_BF16(A, BB, C) \
  __builtin_amdgcn_wmma_f32_16x16x32_bf16(false, (A), false, (BB), (short)0, (C), false, false)

// ---------------- packing kernels ----------------

// h_node_in (B,100,128) f32 -> (B,128,128) bf16, rows 100..127 zeroed
__global__ void pack_hn_kernel(const float* __restrict__ hn,
                               unsigned short* __restrict__ dst) {
  int idx = blockIdx.x * blockDim.x + threadIdx.x;     // B*GP*IND
  int k = idx & 127;
  int g = (idx >> 7) & 127;
  int b = idx >> 14;
  float v = (g < Gq) ? hn[((size_t)b * Gq + g) * IND + k] : 0.f;
  dst[idx] = f2bf(v);
}

// Wt[slot][d][k] = W*[slot][k][d]  (16 slots: 4 Q, 4 K, 8 V), bf16, N-major
// WoT[e][h*32+d] = W_out[h][d][e], bf16, N-major
__global__ void pack_w_kernel(const float* __restrict__ Wq, const float* __restrict__ Wk,
                              const float* __restrict__ Wv, const float* __restrict__ Wo,
                              unsigned short* __restrict__ Wt,
                              unsigned short* __restrict__ WoT) {
  int idx = blockIdx.x * blockDim.x + threadIdx.x;     // 65536
  {
    int k = idx & 127, d = (idx >> 7) & 31, s = idx >> 12;
    const float* src = (s < 4) ? (Wq + (size_t)s * IND * KD)
                     : (s < 8) ? (Wk + (size_t)(s - 4) * IND * KD)
                               : (Wv + (size_t)(s - 8) * IND * KD);
    Wt[idx] = f2bf(src[k * KD + d]);
  }
  if (idx < 128 * 256) {
    int kk = idx & 255, e = idx >> 8;
    int h = kk >> 5, d = kk & 31;
    WoT[idx] = f2bf(Wo[((size_t)h * KD + d) * 128 + e]);
  }
}

// ---------------- GEMM kernels (WMMA) ----------------

// 16 head-slots x B x 8 m-tiles. C tile 16x32, K=128 (4 wmma steps x 2 acc).
// slots 0..3 -> Qb[h][b][g][d], 4..7 -> Kb, 8..15 -> Vt[h][b][d][g] (transposed!)
__global__ void proj_kernel(const unsigned short* __restrict__ hn,
                            const unsigned short* __restrict__ Wt,
                            unsigned short* __restrict__ Qb,
                            unsigned short* __restrict__ Kb,
                            unsigned short* __restrict__ Vt) {
  int wid  = (blockIdx.x * blockDim.x + threadIdx.x) >> 5;
  int lane = threadIdx.x & 31;
  int mt   = wid & 7;
  int b    = (wid >> 3) & 255;
  int slot = wid >> 11;
  const unsigned short* A  = hn + (size_t)b * GP * IND;
  const unsigned short* Bt = Wt + (size_t)slot * KD * IND;
  v8f acc0 = {}, acc1 = {};
  int m0 = mt * 16;
#pragma unroll
  for (int q = 0; q < 4; ++q) {
    int k0 = q * 32;
    v16bf a  = load_a (A,  IND, m0, k0, lane);
    v16bf b0 = load_bt(Bt, IND, 0,  k0, lane);
    v16bf b1 = load_bt(Bt, IND, 16, k0, lane);
    acc0 = WMMA_BF16(a, b0, acc0);
    acc1 = WMMA_BF16(a, b1, acc1);
  }
  int col = lane & 15;
  int rb  = m0 + ((lane >> 4) << 3);
  if (slot < 8) {
    unsigned short* dst = (slot < 4 ? Qb + (size_t)slot * Bq * GP * KD
                                    : Kb + (size_t)(slot - 4) * Bq * GP * KD)
                          + (size_t)b * GP * KD;
#pragma unroll
    for (int v = 0; v < 8; ++v) {
      int g = rb + v;
      dst[(size_t)g * KD + col]      = f2bf(acc0[v]);
      dst[(size_t)g * KD + col + 16] = f2bf(acc1[v]);
    }
  } else {
    unsigned short* dst = Vt + (size_t)(slot - 8) * Bq * KD * GP + (size_t)b * KD * GP;
#pragma unroll
    for (int v = 0; v < 8; ++v) {
      int g = rb + v;
      dst[(size_t)col * GP + g]        = f2bf(acc0[v]);
      dst[(size_t)(col + 16) * GP + g] = f2bf(acc1[v]);
    }
  }
}

// node_comp[h,b,i,j] = NORM * Q[h,b,i,:] . K[h,b,j,:]   (single K=32 wmma / tile)
// written into comp channels 0..3, flat (h,b,i,j) f32 layout, guarded to i,j<100
__global__ void scores_kernel(const unsigned short* __restrict__ Qb,
                              const unsigned short* __restrict__ Kb,
                              float* __restrict__ comp) {
  int wid  = (blockIdx.x * blockDim.x + threadIdx.x) >> 5;
  int lane = threadIdx.x & 31;
  int tj = wid % 7; int t = wid / 7;
  int ti = t % 7;   t /= 7;
  int b  = t & 255; int h = t >> 8;
  const unsigned short* Qh = Qb + ((size_t)h * Bq + b) * GP * KD;
  const unsigned short* Kh = Kb + ((size_t)h * Bq + b) * GP * KD;
  v16bf a  = load_a (Qh, KD, ti * 16, 0, lane);
  v16bf bb = load_bt(Kh, KD, tj * 16, 0, lane);
  v8f acc = {};
  acc = WMMA_BF16(a, bb, acc);
  float* C = comp + ((size_t)h * Bq + b) * Gq * Gq;
  int j  = tj * 16 + (lane & 15);
  int rb = ti * 16 + ((lane >> 4) << 3);
  if (j < Gq) {
#pragma unroll
    for (int v = 0; v < 8; ++v) {
      int i = rb + v;
      if (i < Gq) C[(size_t)i * Gq + j] = NORMQK * acc[v];
    }
  }
}

// ---------------- LSTM scan: one wave32 per batch element ----------------
// comp is flat (h,b,i,j) f32 == flat (s, b, c) with s = 10000 steps, c = 8 chans.
// lane = gate index j in [0,32); gates rows = [i|f|g|o] blocks of 8.
// ys written IN PLACE over comp. The step chain is latency-bound, so each
// wave prefetches its 32B x-slice LSTM_PF steps ahead (global_prefetch_b8).
__global__ void lstm_kernel(float* __restrict__ comp,
                            const float* __restrict__ cost,
                            const float* __restrict__ bcost,
                            const float* __restrict__ hx, const float* __restrict__ cx,
                            const float* __restrict__ Wih, const float* __restrict__ Whh,
                            const float* __restrict__ bih, const float* __restrict__ bhh,
                            float* __restrict__ out_tail) {
  int b = blockIdx.x;
  int lane = threadIdx.x;
  float wih[10], whh[8];
#pragma unroll
  for (int t = 0; t < 10; ++t) wih[t] = Wih[lane * 10 + t];
#pragma unroll
  for (int t = 0; t < 8; ++t)  whh[t] = Whh[lane * 8 + t];
  // cost features are loop-invariant: fold them into the bias once
  float bias = bih[lane] + bhh[lane] + wih[8] * cost[b] + wih[9] * bcost[b];
  float hreg[8];
#pragma unroll
  for (int k = 0; k < 8; ++k) hreg[k] = hx[b * 8 + k];
  float c    = (lane < 8) ? cx[b * 8 + lane] : 0.f;
  float hcur = (lane < 8) ? hreg[lane] : 0.f;
  bool  isG  = (lane >= 16) && (lane < 24);
  float* blk = comp + b * 8;
  // warm the pipeline
#pragma unroll
  for (int p = 0; p < LSTM_PF; p += 4)
    __builtin_prefetch(blk + (size_t)p * 2048, 0, 1);
  for (int s = 0; s < 10000; ++s) {
    const float* x = blk + (size_t)s * 2048;
    if (s + LSTM_PF < 10000)
      __builtin_prefetch(blk + (size_t)(s + LSTM_PF) * 2048, 0, 1);
    float4 x0 = *reinterpret_cast<const float4*>(x);      // 32B-aligned slice
    float4 x1 = *reinterpret_cast<const float4*>(x + 4);
    float g = bias;
    g += wih[0] * x0.x + wih[1] * x0.y + wih[2] * x0.z + wih[3] * x0.w;
    g += wih[4] * x1.x + wih[5] * x1.y + wih[6] * x1.z + wih[7] * x1.w;
#pragma unroll
    for (int k = 0; k < 8; ++k) g += whh[k] * hreg[k];
    float act = isG ? tanhf(g) : 1.f / (1.f + __expf(-g));
    int k8 = lane & 7;
    float ai = __shfl(act, k8,      32);
    float af = __shfl(act, k8 + 8,  32);
    float ag = __shfl(act, k8 + 16, 32);
    float ao = __shfl(act, k8 + 24, 32);
    float cn = af * c + ai * ag;
    float hn = ao * tanhf(cn);
    if (lane < 8) {
      c = cn; hcur = hn;
      blk[(size_t)s * 2048 + lane] = hn;   // ys in place
    }
#pragma unroll
    for (int k = 0; k < 8; ++k) hreg[k] = __shfl(hcur, k, 32);
  }
  if (lane < 8) {
    out_tail[b * 8 + lane]        = hcur;   // h_t
    out_tail[2048 + b * 8 + lane] = c;      // c_t
  }
}

// ---------------- softmax: one wave per (h,b,i) row ----------------
// reads ys (== comp, reinterpreted (8,B,100,100)), writes bf16 attn (8,B,112,128)
// with explicit zero padding so WMMA never touches garbage.
__global__ void softmax_kernel(const float* __restrict__ ys,
                               unsigned short* __restrict__ attn) {
  int wid  = (blockIdx.x * blockDim.x + threadIdx.x) >> 5;
  int lane = threadIdx.x & 31;
  int i = wid % GT; int t = wid / GT;
  int b = t & 255;  int h = t >> 8;
  unsigned short* row = attn + (((size_t)h * Bq + b) * GT + i) * GP;
  if (i >= Gq) {
#pragma unroll
    for (int tt = 0; tt < 4; ++tt) row[lane + 32 * tt] = 0;
    return;
  }
  const float* src = ys + (((size_t)h * Bq + b) * Gq + i) * Gq;
  float v[4], m = -1e30f;
#pragma unroll
  for (int tt = 0; tt < 4; ++tt) {
    int j = lane + 32 * tt;
    v[tt] = (j < Gq) ? src[j] : -1e30f;
    m = fmaxf(m, v[tt]);
  }
#pragma unroll
  for (int off = 16; off > 0; off >>= 1) m = fmaxf(m, __shfl_xor(m, off, 32));
  float e[4], sum = 0.f;
#pragma unroll
  for (int tt = 0; tt < 4; ++tt) {
    int j = lane + 32 * tt;
    e[tt] = (j < Gq) ? __expf(v[tt] - m) : 0.f;
    sum += e[tt];
  }
#pragma unroll
  for (int off = 16; off > 0; off >>= 1) sum += __shfl_xor(sum, off, 32);
  float inv = 1.f / sum;
#pragma unroll
  for (int tt = 0; tt < 4; ++tt) {
    int j = lane + 32 * tt;
    row[j] = (j < Gq) ? f2bf(e[tt] * inv) : (unsigned short)0;
  }
}

// heads[h,b,i,d] = attn(100x128pad) @ V(128pad x 32); stored interleaved as
// heads_cat[b][i][h*32+d] bf16 so the output projection is one K=256 GEMM.
__global__ void attnv_kernel(const unsigned short* __restrict__ attn,
                             const unsigned short* __restrict__ Vt,
                             unsigned short* __restrict__ hc) {
  int wid  = (blockIdx.x * blockDim.x + threadIdx.x) >> 5;
  int lane = threadIdx.x & 31;
  int ti = wid % 7; int t = wid / 7;
  int b  = t & 255; int h = t >> 8;
  const unsigned short* A  = attn + ((size_t)h * Bq + b) * GT * GP;
  const unsigned short* Bt = Vt   + ((size_t)h * Bq + b) * KD * GP;
  v8f acc0 = {}, acc1 = {};
#pragma unroll
  for (int q = 0; q < 4; ++q) {
    int k0 = q * 32;
    v16bf a  = load_a (A,  GP, ti * 16, k0, lane);
    v16bf b0 = load_bt(Bt, GP, 0,  k0, lane);
    v16bf b1 = load_bt(Bt, GP, 16, k0, lane);
    acc0 = WMMA_BF16(a, b0, acc0);
    acc1 = WMMA_BF16(a, b1, acc1);
  }
  unsigned short* dst = hc + (size_t)b * GT * 256 + h * 32;
  int col = lane & 15;
  int rb  = ti * 16 + ((lane >> 4) << 3);
#pragma unroll
  for (int v = 0; v < 8; ++v) {
    int i = rb + v;
    dst[(size_t)i * 256 + col]      = f2bf(acc0[v]);
    dst[(size_t)i * 256 + col + 16] = f2bf(acc1[v]);
  }
}

// out[b,i,e] = heads_cat(112x256) @ WoT^T(256x128), f32 result, i<100 stored.
__global__ void out_kernel(const unsigned short* __restrict__ hc,
                           const unsigned short* __restrict__ WoT,
                           float* __restrict__ out) {
  int wid  = (blockIdx.x * blockDim.x + threadIdx.x) >> 5;
  int lane = threadIdx.x & 31;
  int nt = wid & 3; int t = wid >> 2;
  int ti = t % 7;   int b = t / 7;
  const unsigned short* A = hc + (size_t)b * GT * 256;
  v8f acc0 = {}, acc1 = {};
#pragma unroll
  for (int q = 0; q < 8; ++q) {
    int k0 = q * 32;
    v16bf a  = load_a (A,   256, ti * 16, k0, lane);
    v16bf b0 = load_bt(WoT, 256, nt * 32,      k0, lane);
    v16bf b1 = load_bt(WoT, 256, nt * 32 + 16, k0, lane);
    acc0 = WMMA_BF16(a, b0, acc0);
    acc1 = WMMA_BF16(a, b1, acc1);
  }
  int e  = nt * 32 + (lane & 15);
  int rb = ti * 16 + ((lane >> 4) << 3);
#pragma unroll
  for (int v = 0; v < 8; ++v) {
    int i = rb + v;
    if (i < Gq) {
      out[((size_t)b * Gq + i) * 128 + e]      = acc0[v];
      out[((size_t)b * Gq + i) * 128 + e + 16] = acc1[v];
    }
  }
}

// ---------------- launcher ----------------

extern "C" void kernel_launch(void* const* d_in, const int* in_sizes, int n_in,
                              void* d_out, int out_size, void* d_ws, size_t ws_size,
                              hipStream_t stream) {
  const float* h_node  = (const float*)d_in[0];
  const float* pos     = (const float*)d_in[1];
  const float* bpos    = (const float*)d_in[2];
  const float* cost    = (const float*)d_in[3];
  const float* bcost   = (const float*)d_in[4];
  const float* hx      = (const float*)d_in[5];
  const float* cx      = (const float*)d_in[6];
  const float* Wq      = (const float*)d_in[7];
  const float* Wk      = (const float*)d_in[8];
  const float* Wv      = (const float*)d_in[9];
  const float* Wo      = (const float*)d_in[10];
  const float* Wih     = (const float*)d_in[11];
  const float* Whh     = (const float*)d_in[12];
  const float* bih     = (const float*)d_in[13];
  const float* bhh     = (const float*)d_in[14];
  float* out = (float*)d_out;

  char* w = (char*)d_ws;
  size_t off = 0;
  float*          comp = (float*)(w + off);          off += 81920000;   // (8,B,100,100) f32
  unsigned short* hnb  = (unsigned short*)(w + off); off += 8388608;    // (B,128,128) bf16
  unsigned short* Qb   = (unsigned short*)(w + off); off += 8388608;    // (4,B,128,32)
  unsigned short* Kb   = (unsigned short*)(w + off); off += 8388608;    // (4,B,128,32)
  unsigned short* Vt   = (unsigned short*)(w + off); off += 16777216;   // (8,B,32,128)
  unsigned short* attn = (unsigned short*)(w + off); off += 58720256;   // (8,B,112,128)
  unsigned short* hc   = (unsigned short*)(w + off); off += 14680064;   // (B,112,256)
  unsigned short* Wt   = (unsigned short*)(w + off); off += 131072;     // (16,32,128)
  unsigned short* WoT  = (unsigned short*)(w + off); off += 65536;      // (128,256)

  pack_hn_kernel<<<16384, 256, 0, stream>>>(h_node, hnb);
  pack_w_kernel<<<256, 256, 0, stream>>>(Wq, Wk, Wv, Wo, Wt, WoT);
  proj_kernel<<<4096, 256, 0, stream>>>(hnb, Wt, Qb, Kb, Vt);            // 32768 waves
  scores_kernel<<<6272, 256, 0, stream>>>(Qb, Kb, comp);                 // 50176 waves

  // channels 4..7 of comp are the pos / best_pos tensors, contiguous in this layout
  hipMemcpyAsync(comp + 10240000, pos,  20480000, hipMemcpyDeviceToDevice, stream);
  hipMemcpyAsync(comp + 15360000, bpos, 20480000, hipMemcpyDeviceToDevice, stream);

  lstm_kernel<<<Bq, 32, 0, stream>>>(comp, cost, bcost, hx, cx,
                                     Wih, Whh, bih, bhh, out + 3276800);
  softmax_kernel<<<28672, 256, 0, stream>>>(comp, attn);                 // 229376 waves
  attnv_kernel<<<1792, 256, 0, stream>>>(attn, Vt, hc);                  // 14336 waves
  out_kernel<<<896, 256, 0, stream>>>(hc, WoT, out);                     // 7168 waves
}
